// SNN_36335423324556
// MI455X (gfx1250) — compile-verified, matrix operand order checked by reading, not run on hardware
//
#include <hip/hip_runtime.h>

#define T_STEPS 100
#define BATCH   4096
#define D_IN    784
#define D_INP   800       // D_IN padded to a multiple of 32 (bf16 weight copy)
#define D_H     1024
#define D_OUT   26
#define BETA    0.8f
#define THRESH  1.0f

typedef __bf16 bf16;
typedef bf16  v16bf __attribute__((ext_vector_type(16)));
typedef float v8f   __attribute__((ext_vector_type(8)));

// ---- helpers -------------------------------------------------------------

// round-to-nearest-even f32 -> bf16 bits kept in the high half of a dword
__device__ __forceinline__ unsigned rne_hi(float a) {
  union { float f; unsigned u; } x; x.f = a;
  return x.u + 0x7FFFu + ((x.u >> 16) & 1u);
}

// pack two f32 into two bf16 (RNE) with a single v_perm_b32 for the pack:
// result bytes [1:0] = hi16(rne(a)), bytes [3:2] = hi16(rne(b))
__device__ __forceinline__ unsigned pk2(float a, float b) {
  return __builtin_amdgcn_perm(rne_hi(b), rne_hi(a), 0x07060302u);
}

// 8 consecutive f32 -> 8 bf16 (16 bytes), unconditional
__device__ __forceinline__ uint4 cvt8f(const float* __restrict__ row, int kb) {
  float4 f0 = *(const float4*)(row + kb);
  float4 f1 = *(const float4*)(row + kb + 4);
  uint4 r;
  r.x = pk2(f0.x, f0.y); r.y = pk2(f0.z, f0.w);
  r.z = pk2(f1.x, f1.y); r.w = pk2(f1.z, f1.w);
  return r;
}

// guarded variant for the final K-tile (784 is a multiple of 8 -> chunks are
// all-valid or all-invalid); used exactly once per block
__device__ __forceinline__ uint4 cvt8g(const float* __restrict__ row, int kb) {
  if (kb + 8 <= D_IN) return cvt8f(row, kb);
  uint4 z; z.x = 0u; z.y = 0u; z.z = 0u; z.w = 0u;
  return z;
}

// Load a 16x32 bf16 WMMA operand fragment from an LDS tile stored as
// [16 rows][32 K] (64B rows). Per ISA 16-bit A/B layout:
//   lanes 0-15 : K 0-7 and K 16-23, row = lane
//   lanes 16-31: K 8-15 and K 24-31, row = lane-16
__device__ __forceinline__ v16bf ld_frag(const bf16* tile, int lrow, int lhi) {
  const bf16* p = tile + lrow * 32 + lhi * 8;
  uint4 lo = *(const uint4*)p;
  uint4 hi = *(const uint4*)(p + 16);
  v16bf f;
  __builtin_memcpy(&f, &lo, 16);
  __builtin_memcpy(((char*)&f) + 16, &hi, 16);
  return f;
}

// ---- one-time prep kernels ----------------------------------------------

__global__ void snn_zero(float* __restrict__ p, int n) {
  int i = blockIdx.x * blockDim.x + threadIdx.x;
  const int stride = gridDim.x * blockDim.x;
  for (; i < n; i += stride) p[i] = 0.f;
}

// W1 (1024 x 784 f32) -> bf16 [1024][800], zero-padded K
__global__ void snn_cvt_w1(const float* __restrict__ W1, unsigned short* __restrict__ W1b) {
  int i = blockIdx.x * blockDim.x + threadIdx.x;
  if (i >= D_H * D_INP) return;
  const int row = i / D_INP, col = i - row * D_INP;
  unsigned short v = 0;
  if (col < D_IN) v = (unsigned short)(rne_hi(W1[(size_t)row * D_IN + col]) >> 16);
  W1b[i] = v;
}

// W2 (26 x 1024 f32) -> bf16 [32][1024], zero-padded rows
__global__ void snn_cvt_w2(const float* __restrict__ W2, unsigned short* __restrict__ W2b) {
  int i = blockIdx.x * blockDim.x + threadIdx.x;
  if (i >= 32 * D_H) return;
  const int row = i / D_H, col = i - row * D_H;
  unsigned short v = 0;
  if (row < D_OUT) v = (unsigned short)(rne_hi(W2[(size_t)row * D_H + col]) >> 16);
  W2b[i] = v;
}

// ---- layer 1: cur1 = x_t @ W1^T + b1, fused leaky-integrate-fire ---------
// block tile 128(M) x 128(N), 8 waves 4(M) x 2(N), 2x4 accumulators/wave.

__global__ __launch_bounds__(256) void snn_l1(const float* __restrict__ X,
                                              const unsigned short* __restrict__ W1b,
                                              const float* __restrict__ b1,
                                              float* __restrict__ mem1,
                                              unsigned short* __restrict__ spk1raw)
{
  __shared__ __align__(16) bf16 lA[2][128][32];   // 16 KB
  __shared__ __align__(16) bf16 lB[2][128][32];   // 16 KB

  const int tid  = threadIdx.x;
  const int lane = tid & 31;
  const int wave = tid >> 5;
  const int wm = wave >> 1, wn = wave & 1;
  const int lrow = lane & 15, lhi = lane >> 4;
  const int bm = blockIdx.y * 128, bn = blockIdx.x * 128;

  // staging: 256 threads fill 128 rows x 64B per tile -> 2 threads/row, 32B each
  const int srow = tid >> 1;
  const int se0  = (tid & 1) * 16;
  const float*          xrow  = X   + (size_t)(bm + srow) * D_IN;
  const unsigned short* wrowb = W1b + (size_t)(bn + srow) * D_INP;

  v8f acc[2][4];
  #pragma unroll
  for (int i = 0; i < 2; ++i)
    #pragma unroll
    for (int j = 0; j < 4; ++j)
      #pragma unroll
      for (int r = 0; r < 8; ++r) acc[i][j][r] = 0.f;

  const int NK = D_INP / 32;                       // 25 (last x tile zero-padded)

  // prologue: tile 0 (fully in-range)
  uint4 a0 = cvt8f(xrow, se0), a1 = cvt8f(xrow, se0 + 8);
  uint4 c0 = *(const uint4*)(wrowb + se0);
  uint4 c1 = *(const uint4*)(wrowb + se0 + 8);
  *(uint4*)&lA[0][srow][se0]     = a0;
  *(uint4*)&lA[0][srow][se0 + 8] = a1;
  *(uint4*)&lB[0][srow][se0]     = c0;
  *(uint4*)&lB[0][srow][se0 + 8] = c1;
  __syncthreads();

  for (int kt = 0; kt < NK; ++kt) {
    const int  cur  = kt & 1;
    const bool more = (kt + 1) < NK;
    if (more) {                                    // global prefetch for next tile
      const int kb = (kt + 1) * 32 + se0;
      c0 = *(const uint4*)(wrowb + kb);            // weights: pure bf16 copy
      c1 = *(const uint4*)(wrowb + kb + 8);
      if (kt + 1 < NK - 1) {                       // uniform branch: full tiles
        a0 = cvt8f(xrow, kb); a1 = cvt8f(xrow, kb + 8);
      } else {                                     // only once: edge tile
        a0 = cvt8g(xrow, kb); a1 = cvt8g(xrow, kb + 8);
      }
    }
    v16bf af[2], bfr[4];
    #pragma unroll
    for (int mi = 0; mi < 2; ++mi)
      af[mi] = ld_frag(&lA[cur][wm * 32 + mi * 16][0], lrow, lhi);
    #pragma unroll
    for (int ni = 0; ni < 4; ++ni)
      bfr[ni] = ld_frag(&lB[cur][wn * 64 + ni * 16][0], lrow, lhi);
    #pragma unroll
    for (int mi = 0; mi < 2; ++mi)
      #pragma unroll
      for (int ni = 0; ni < 4; ++ni)
        acc[mi][ni] = __builtin_amdgcn_wmma_f32_16x16x32_bf16(
            false, af[mi], false, bfr[ni], (short)0, acc[mi][ni], false, false);
    if (more) {
      const int nxt = cur ^ 1;
      *(uint4*)&lA[nxt][srow][se0]     = a0;
      *(uint4*)&lA[nxt][srow][se0 + 8] = a1;
      *(uint4*)&lB[nxt][srow][se0]     = c0;
      *(uint4*)&lB[nxt][srow][se0 + 8] = c1;
    }
    __syncthreads();
  }

  // fused LIF epilogue: one RMW pass over mem1, spikes emitted as bf16 0/1
  #pragma unroll
  for (int ni = 0; ni < 4; ++ni) {
    const int n = bn + wn * 64 + ni * 16 + lrow;
    const float bias = b1[n];
    #pragma unroll
    for (int mi = 0; mi < 2; ++mi) {
      const int mb = bm + wm * 32 + mi * 16 + lhi * 8;
      #pragma unroll
      for (int r = 0; r < 8; ++r) {
        const int m = mb + r;
        const size_t idx = (size_t)m * D_H + n;
        const float curI = acc[mi][ni][r] + bias;
        const float old  = mem1[idx];
        const float rst  = (old > THRESH) ? THRESH : 0.f;
        const float memv = BETA * old + curI - rst;
        mem1[idx] = memv;
        spk1raw[idx] = (memv > THRESH) ? (unsigned short)0x3F80u   // bf16 1.0
                                       : (unsigned short)0u;       // bf16 0.0
      }
    }
  }
}

// ---- layer 2: cur2 = spk1 @ W2^T + b2, fused LIF + per-step outputs ------
// block tile 64(M) x 32(N); 8 waves 4(M) x 2(N), one accumulator per wave.

__global__ __launch_bounds__(256) void snn_l2(const unsigned short* __restrict__ spk1raw,
                                              const unsigned short* __restrict__ W2b,
                                              const float* __restrict__ b2,
                                              float* __restrict__ mem2,
                                              float* __restrict__ out_spk,
                                              float* __restrict__ out_mem)
{
  __shared__ __align__(16) bf16 lA[2][64][32];    // 8 KB
  __shared__ __align__(16) bf16 lB[2][32][32];    // 4 KB

  const int tid  = threadIdx.x;
  const int lane = tid & 31;
  const int wave = tid >> 5;
  const int wm = wave >> 1, wn = wave & 1;
  const int lrow = lane & 15, lhi = lane >> 4;
  const int bm = blockIdx.x * 64;

  // A staging: 64 rows x 64B -> 4 threads/row, 16B each (already bf16)
  const int arow = tid >> 2;
  const int ae0  = (tid & 3) * 8;
  const unsigned short* asrc = spk1raw + (size_t)(bm + arow) * D_H;
  // B staging: threads 0..127 copy 32 x 64B from the padded bf16 W2 copy
  const bool bact = tid < 128;
  const int brow = tid >> 2;                       // 0..31 when bact
  const int be0  = (tid & 3) * 8;
  const unsigned short* wrowb = W2b + (size_t)brow * D_H;

  v8f acc;
  #pragma unroll
  for (int r = 0; r < 8; ++r) acc[r] = 0.f;

  uint4 a = *(const uint4*)(asrc + ae0);
  uint4 b = make_uint4(0u, 0u, 0u, 0u);
  if (bact) b = *(const uint4*)(wrowb + be0);
  *(uint4*)&lA[0][arow][ae0] = a;
  if (bact) *(uint4*)&lB[0][brow][be0] = b;
  __syncthreads();

  const int NK = D_H / 32;                          // 32
  for (int kt = 0; kt < NK; ++kt) {
    const int  cur  = kt & 1;
    const bool more = (kt + 1) < NK;
    if (more) {
      a = *(const uint4*)(asrc + (kt + 1) * 32 + ae0);
      if (bact) b = *(const uint4*)(wrowb + (kt + 1) * 32 + be0);
    }
    v16bf af  = ld_frag(&lA[cur][wm * 16][0], lrow, lhi);
    v16bf bfr = ld_frag(&lB[cur][wn * 16][0], lrow, lhi);
    acc = __builtin_amdgcn_wmma_f32_16x16x32_bf16(
        false, af, false, bfr, (short)0, acc, false, false);
    if (more) {
      const int nxt = cur ^ 1;
      *(uint4*)&lA[nxt][arow][ae0] = a;
      if (bact) *(uint4*)&lB[nxt][brow][be0] = b;
    }
    __syncthreads();
  }

  // fused LIF + output writes (only the real 26 columns)
  const int n = wn * 16 + lrow;
  if (n < D_OUT) {
    const float bias = b2[n];
    const int mb = bm + wm * 16 + lhi * 8;
    #pragma unroll
    for (int r = 0; r < 8; ++r) {
      const int m = mb + r;
      const size_t i2 = (size_t)m * 32 + n;         // mem2 padded to 32 cols
      const float curI = acc[r] + bias;
      const float old  = mem2[i2];
      const float rst  = (old > THRESH) ? THRESH : 0.f;
      const float memv = BETA * old + curI - rst;
      mem2[i2] = memv;
      const size_t o = (size_t)m * D_OUT + n;
      out_spk[o] = (memv > THRESH) ? 1.f : 0.f;
      out_mem[o] = memv;
    }
  }
}

// ---- host launch ---------------------------------------------------------

extern "C" void kernel_launch(void* const* d_in, const int* in_sizes, int n_in,
                              void* d_out, int out_size, void* d_ws, size_t ws_size,
                              hipStream_t stream) {
  (void)in_sizes; (void)n_in; (void)out_size; (void)ws_size;
  const float* x  = (const float*)d_in[0];
  const float* W1 = (const float*)d_in[1];
  const float* b1 = (const float*)d_in[2];
  const float* W2 = (const float*)d_in[3];
  const float* b2 = (const float*)d_in[4];
  float* out = (float*)d_out;

  // workspace layout (27.4 MB):
  //   spk1 bf16 | mem1 f32 | mem2 f32(pad 32) | W1 bf16 [1024][800] | W2 bf16 [32][1024]
  char* ws = (char*)d_ws;
  const size_t off_mem1 = (size_t)BATCH * D_H * 2;                 //  8,388,608
  const size_t off_mem2 = off_mem1 + (size_t)BATCH * D_H * 4;      // 25,165,824
  const size_t off_w1b  = off_mem2 + (size_t)BATCH * 32 * 4;       // 25,690,112
  const size_t off_w2b  = off_w1b + (size_t)D_H * D_INP * 2;       // 27,328,512
  unsigned short* spk1 = (unsigned short*)ws;
  float*          mem1 = (float*)(ws + off_mem1);
  unsigned short* W1b  = (unsigned short*)(ws + off_w1b);
  unsigned short* W2b  = (unsigned short*)(ws + off_w2b);
  float*          mem2 = (float*)(ws + off_mem2);

  // one-time per call: weight bf16 copies + zero the membrane state
  snn_cvt_w1<<<(D_H * D_INP + 255) / 256, 256, 0, stream>>>(W1, W1b);
  snn_cvt_w2<<<(32 * D_H + 255) / 256, 256, 0, stream>>>(W2, W2b);
  snn_zero<<<2048, 256, 0, stream>>>(mem1, BATCH * D_H + BATCH * 32);

  const size_t out_half = (size_t)T_STEPS * BATCH * D_OUT;
  for (int t = 0; t < T_STEPS; ++t) {
    const float* Xt = x + (size_t)t * BATCH * D_IN;
    snn_l1<<<dim3(D_H / 128, BATCH / 128), 256, 0, stream>>>(Xt, W1b, b1, mem1, spk1);
    snn_l2<<<dim3(BATCH / 64), 256, 0, stream>>>(spk1, W2b, b2, mem2,
        out + (size_t)t * BATCH * D_OUT,
        out + out_half + (size_t)t * BATCH * D_OUT);
  }
}
